// Seq2Seq_13082470384182
// MI455X (gfx1250) — compile-verified, matrix operand order checked by reading, not run on hardware
//
#include <hip/hip_runtime.h>
#include <hip/hip_bf16.h>
#include <math.h>

// Seq2Seq LSTM + attention for MI455X (gfx1250, wave32).
// Dominant cost: 48x [64 x 32000 x 1024] logits GEMM -> bf16 WMMA
// (v_wmma_f32_16x16x32_bf16), W_y converted once to bf16 (65.5MB, L2-resident).
// GEMM waves block 4 n-tiles and reuse the A fragment across 4 WMMAs per k-step.

#define V 32000
#define Hd 512
#define E 512
#define Bb 64
#define Ss 128
#define Tt 48

typedef __attribute__((ext_vector_type(16))) __bf16       v16bf;
typedef __attribute__((ext_vector_type(8)))  float        v8f;
typedef __attribute__((ext_vector_type(4)))  unsigned int u32x4;

union V16B { v16bf v; u32x4 q[2]; };

// ---- CDNA5 16-bit A/B fragment load (16x32 tile), per ISA 7.12.2:
// lane<16: K = k0+0..7 and k0+16..23 ; lane>=16: K = k0+8..15 and k0+24..31
__device__ __forceinline__ v16bf load_frag_row(const __hip_bfloat16* prow, int k0, int lane) {
    int ksel = (lane >> 4) & 1;
    const __hip_bfloat16* p = prow + k0 + ksel * 8;
    V16B t;
    t.q[0] = *(const u32x4*)(p);
    t.q[1] = *(const u32x4*)(p + 16);
    return t.v;
}

// A matrix is [64 x 1024] split into two [64 x 512] bf16 segments (concat in K).
__device__ __forceinline__ v16bf load_a_split(const __hip_bfloat16* seg0,
                                              const __hip_bfloat16* seg1,
                                              int m0, int k0, int lane) {
    int r = m0 + (lane & 15);
    if (k0 < 512) return load_frag_row(seg0 + (size_t)r * 512, k0, lane);
    return load_frag_row(seg1 + (size_t)r * 512, k0 - 512, lane);
}

// B = W^T : column n of B is row n of row-major W [N x 1024]
__device__ __forceinline__ v16bf load_b_w(const __hip_bfloat16* w, int n0, int k0, int lane) {
    int r = n0 + (lane & 15);
    return load_frag_row(w + (size_t)r * 1024, k0, lane);
}

__device__ __forceinline__ v8f wmma_bf16(v16bf a, v16bf b, v8f c) {
    return __builtin_amdgcn_wmma_f32_16x16x32_bf16(false, a, false, b, (short)0, c, false, false);
}

__device__ __forceinline__ float sigmoidf(float x) { return 1.f / (1.f + __expf(-x)); }

// ---------------- setup kernels ----------------

// Stable descending argsort of 64 lengths (replicates jnp.argsort(-lengths)).
__global__ void k_sort(const int* lens, int* order, int* lens_s) {
    __shared__ int L[Bb];
    int i = threadIdx.x;
    L[i] = lens[i];
    __syncthreads();
    int li = L[i], rank = 0;
    for (int j = 0; j < Bb; ++j) {
        int lj = L[j];
        if (lj > li || (lj == li && j < i)) rank++;
    }
    order[rank] = i;
    lens_s[rank] = li;
}

__global__ void k_cvt_bf16(const float* src, __hip_bfloat16* dst, int n) {
    for (int i = blockIdx.x * blockDim.x + threadIdx.x; i < n; i += gridDim.x * blockDim.x)
        dst[i] = __float2bfloat16(src[i]);
}

// concat [2048 x 512] | [2048 x 512] -> bf16 [2048 x 1024]
__global__ void k_cat_w(const float* wih, const float* whh, __hip_bfloat16* dst) {
    const int total = 2048 * 1024;
    for (int i = blockIdx.x * blockDim.x + threadIdx.x; i < total; i += gridDim.x * blockDim.x) {
        int n = i >> 10, k = i & 1023;
        float v = (k < 512) ? wih[n * 512 + k] : whh[n * 512 + (k - 512)];
        dst[i] = __float2bfloat16(v);
    }
}

// gather emb rows (sorted batch order) -> bf16 [seqlen, 64, 512]
__global__ void k_gather(const float* emb, const int* tok, const int* order,
                         int seqlen, __hip_bfloat16* dst) {
    const int total = seqlen * Bb * 512;
    for (int i = blockIdx.x * blockDim.x + threadIdx.x; i < total; i += gridDim.x * blockDim.x) {
        int e = i & 511;
        int b = (i >> 9) & 63;
        int s = i >> 15;
        int row = tok[order[b] * seqlen + s];
        dst[i] = __float2bfloat16(emb[(size_t)row * 512 + e]);
    }
}

__global__ void k_zero(float* p, int n) {
    for (int i = blockIdx.x * blockDim.x + threadIdx.x; i < n; i += gridDim.x * blockDim.x)
        p[i] = 0.f;
}

// ---------------- fused LSTM step (encoder & decoder) ----------------
// gates[64,2048] = A @ W^T via WMMA; each wave owns the i/f/g/o tiles of one
// (16 rows x 16 cols) patch so the nonlinearity is applied in-wave.
// grid = 16 blocks (j0 = blk*32), block = 256 (8 waves).
__global__ void __launch_bounds__(256)
k_lstm_step(const __hip_bfloat16* a0, const __hip_bfloat16* a1,  // A segments [64,512]
            const __hip_bfloat16* W,                             // [2048,1024] bf16
            const float* hprev, const float* cprev,
            float* hnew, float* cnew, __hip_bfloat16* hnewb,
            float* hs_out, const int* lens, int step) {          // lens==null -> decoder
    int lane  = threadIdx.x & 31;
    int wv    = threadIdx.x >> 5;      // 0..7
    int mtile = wv & 3;                // 0..3 -> batch rows
    int ntile = wv >> 2;               // 0..1
    int m0 = mtile * 16;
    int j0 = blockIdx.x * 32 + ntile * 16;   // j in [0,512)

    v8f acc[4] = {};
    #pragma unroll 2
    for (int k0 = 0; k0 < 1024; k0 += 32) {
        v16bf a = load_a_split(a0, a1, m0, k0, lane);
        #pragma unroll
        for (int g = 0; g < 4; ++g) {
            v16bf b = load_b_w(W, g * 512 + j0, k0, lane);
            acc[g] = wmma_bf16(a, b, acc[g]);
        }
    }

    int n  = lane & 15;
    int mh = (lane >> 4) & 1;
    int j  = j0 + n;
    #pragma unroll
    for (int i = 0; i < 8; ++i) {
        int b  = m0 + mh * 8 + i;                 // sorted batch index
        float ig = acc[0][i], fg = acc[1][i], gg = acc[2][i], og = acc[3][i];
        float cp = cprev[b * 512 + j];
        float cn = sigmoidf(fg) * cp + sigmoidf(ig) * tanhf(gg);
        float hn = sigmoidf(og) * tanhf(cn);
        if (lens) {  // encoder: freeze past length, zero-pad hs
            bool valid = step < lens[b];
            float hp = hprev[b * 512 + j];
            float h2 = valid ? hn : hp;
            float c2 = valid ? cn : cp;
            hnew[b * 512 + j] = h2;
            cnew[b * 512 + j] = c2;
            hnewb[b * 512 + j] = __float2bfloat16(h2);
            hs_out[((size_t)b * Ss + step) * 512 + j] = valid ? hn : 0.f;
        } else {     // decoder
            hnew[b * 512 + j] = hn;
            cnew[b * 512 + j] = cn;
            hnewb[b * 512 + j] = __float2bfloat16(hn);
        }
    }
}

// ---------------- generic WMMA GEMMs (4 n-tiles per wave, shared A frag) ----
// wave job gid: mtile = gid & 3, ngroup = gid >> 2 covers n0 = ngroup*64 .. +63

__global__ void __launch_bounds__(256)
k_gemm_to_bf16(const __hip_bfloat16* a0, const __hip_bfloat16* a1,
               const __hip_bfloat16* W, __hip_bfloat16* out, int ncols) {
    int lane = threadIdx.x & 31;
    int gid  = blockIdx.x * 8 + (threadIdx.x >> 5);
    int m0 = (gid & 3) * 16;
    int n0 = (gid >> 2) * 64;
    v8f acc[4] = {};
    #pragma unroll 2
    for (int k0 = 0; k0 < 1024; k0 += 32) {
        v16bf a = load_a_split(a0, a1, m0, k0, lane);
        #pragma unroll
        for (int nt = 0; nt < 4; ++nt) {
            v16bf b = load_b_w(W, n0 + nt * 16, k0, lane);
            acc[nt] = wmma_bf16(a, b, acc[nt]);
        }
    }
    int n = lane & 15, mh = (lane >> 4) & 1;
    #pragma unroll
    for (int nt = 0; nt < 4; ++nt)
        #pragma unroll
        for (int i = 0; i < 8; ++i)
            out[(size_t)(m0 + mh * 8 + i) * ncols + n0 + nt * 16 + n] =
                __float2bfloat16(acc[nt][i]);
}

// f32 output; for logits: 4 m-tiles x 500 n-groups = 2000 waves = 250 blocks
__global__ void __launch_bounds__(256)
k_gemm_to_f32(const __hip_bfloat16* a0, const __hip_bfloat16* a1,
              const __hip_bfloat16* W, float* out, int ldo) {
    int lane = threadIdx.x & 31;
    int gid  = blockIdx.x * 8 + (threadIdx.x >> 5);
    int m0 = (gid & 3) * 16;
    int n0 = (gid >> 2) * 64;
    v8f acc[4] = {};
    #pragma unroll 2
    for (int k0 = 0; k0 < 1024; k0 += 32) {
        v16bf a = load_a_split(a0, a1, m0, k0, lane);
        #pragma unroll
        for (int nt = 0; nt < 4; ++nt) {
            v16bf b = load_b_w(W, n0 + nt * 16, k0, lane);
            acc[nt] = wmma_bf16(a, b, acc[nt]);
        }
    }
    int n = lane & 15, mh = (lane >> 4) & 1;
    #pragma unroll
    for (int nt = 0; nt < 4; ++nt)
        #pragma unroll
        for (int i = 0; i < 8; ++i)
            out[(size_t)(m0 + mh * 8 + i) * ldo + n0 + nt * 16 + n] = acc[nt][i];
}

// ---------------- attention pieces ----------------
__global__ void k_att1(const float* hs, const float* Wa1, float* att1) {
    int idx = blockIdx.x * blockDim.x + threadIdx.x;  // 0..8191 = (bs,s)
    if (idx >= Bb * Ss) return;
    const float* row = hs + (size_t)idx * 512;
    float acc = 0.f;
    for (int k = 0; k < 512; ++k) acc += row[k] * Wa1[k];
    att1[idx] = tanhf(acc);
}

// one block per sorted batch row: ta2 scalar, masked softmax over S, ctx.
// NOTE: mask uses UNSORTED mr_lengths indexed at sorted position (reference bug).
__global__ void k_attn(const float* sprev, const float* Wa2, const float* att1,
                       const float* hs, const int* lens_raw,
                       float* ctx, __hip_bfloat16* ctxb) {
    int bs = blockIdx.x, tid = threadIdx.x;  // 256 threads
    __shared__ float red[256];
    __shared__ float alpha[Ss];

    float a = 0.f;
    for (int k = tid; k < 512; k += 256) a += sprev[bs * 512 + k] * Wa2[k];
    red[tid] = a; __syncthreads();
    for (int off = 128; off; off >>= 1) { if (tid < off) red[tid] += red[tid + off]; __syncthreads(); }
    float ta2 = tanhf(red[0]);
    __syncthreads();

    int len = lens_raw[bs];
    float e = -1e30f;
    if (tid < Ss) e = (tid < len) ? att1[bs * Ss + tid] * ta2 : -1e9f;
    red[tid] = e; __syncthreads();
    for (int off = 128; off; off >>= 1) { if (tid < off) red[tid] = fmaxf(red[tid], red[tid + off]); __syncthreads(); }
    float mx = red[0]; __syncthreads();
    float ex = 0.f;
    if (tid < Ss) { ex = __expf(e - mx); alpha[tid] = ex; }
    red[tid] = ex; __syncthreads();
    for (int off = 128; off; off >>= 1) { if (tid < off) red[tid] += red[tid + off]; __syncthreads(); }
    float inv = 1.f / red[0]; __syncthreads();
    if (tid < Ss) alpha[tid] *= inv;
    __syncthreads();

    for (int h = tid; h < 512; h += 256) {
        const float* hp = hs + (size_t)bs * Ss * 512 + h;
        float acc = 0.f;
        #pragma unroll 4
        for (int s = 0; s < Ss; ++s) acc += alpha[s] * hp[s * 512];
        ctx[bs * 512 + h] = acc;
        ctxb[bs * 512 + h] = __float2bfloat16(acc);
    }
}

// softmax over V, write to output row order[bs] (desort) at step t
__global__ void k_softmax_out(const float* logits, const int* order, float* out, int t) {
    int bs = blockIdx.x, tid = threadIdx.x;  // 256 threads
    const float* row = logits + (size_t)bs * V;
    __shared__ float red[256];
    float mx = -1e30f;
    for (int v = tid; v < V; v += 256) mx = fmaxf(mx, row[v]);
    red[tid] = mx; __syncthreads();
    for (int off = 128; off; off >>= 1) { if (tid < off) red[tid] = fmaxf(red[tid], red[tid + off]); __syncthreads(); }
    mx = red[0]; __syncthreads();
    float sm = 0.f;
    for (int v = tid; v < V; v += 256) sm += __expf(row[v] - mx);
    red[tid] = sm; __syncthreads();
    for (int off = 128; off; off >>= 1) { if (tid < off) red[tid] += red[tid + off]; __syncthreads(); }
    float inv = 1.f / red[0];
    float* orow = out + ((size_t)order[bs] * Tt + t) * V;
    for (int v = tid; v < V; v += 256) orow[v] = __expf(row[v] - mx) * inv;
}

// ---------------- host ----------------
extern "C" void kernel_launch(void* const* d_in, const int* in_sizes, int n_in,
                              void* d_out, int out_size, void* d_ws, size_t ws_size,
                              hipStream_t stream) {
    const int*   mr         = (const int*)d_in[0];
    const int*   ref        = (const int*)d_in[1];
    const int*   mr_lengths = (const int*)d_in[2];
    const float* emb        = (const float*)d_in[3];
    const float* W_ih_enc   = (const float*)d_in[4];
    const float* W_hh_enc   = (const float*)d_in[5];
    const float* W_ih_dec   = (const float*)d_in[6];
    const float* W_hh_dec   = (const float*)d_in[7];
    const float* W_y        = (const float*)d_in[8];
    const float* W_s        = (const float*)d_in[9];
    const float* W_a1       = (const float*)d_in[10];
    const float* W_a2       = (const float*)d_in[11];
    float* out = (float*)d_out;

    char* ws = (char*)d_ws;
    size_t off = 0;
    auto alloc = [&](size_t bytes) -> char* {
        char* p = ws + off;
        off = (off + bytes + 255) & ~(size_t)255;
        return p;
    };
    __hip_bfloat16* Wenc  = (__hip_bfloat16*)alloc((size_t)2048 * 1024 * 2);
    __hip_bfloat16* Wdec  = (__hip_bfloat16*)alloc((size_t)2048 * 1024 * 2);
    __hip_bfloat16* Wsb   = (__hip_bfloat16*)alloc((size_t)512 * 1024 * 2);
    __hip_bfloat16* Wyb   = (__hip_bfloat16*)alloc((size_t)V * 1024 * 2);      // 65.5MB, L2-resident
    __hip_bfloat16* Xb    = (__hip_bfloat16*)alloc((size_t)Ss * Bb * 512 * 2);
    __hip_bfloat16* Yb    = (__hip_bfloat16*)alloc((size_t)Tt * Bb * 512 * 2);
    float*          hs    = (float*)alloc((size_t)Bb * Ss * 512 * 4);
    float*          att1  = (float*)alloc((size_t)Bb * Ss * 4);
    float*          hbuf  = (float*)alloc((size_t)2 * Bb * 512 * 4);
    float*          cbuf  = (float*)alloc((size_t)2 * Bb * 512 * 4);
    __hip_bfloat16* hbufb = (__hip_bfloat16*)alloc((size_t)2 * Bb * 512 * 2);
    float*          ctx   = (float*)alloc((size_t)Bb * 512 * 4);
    __hip_bfloat16* ctxb  = (__hip_bfloat16*)alloc((size_t)Bb * 512 * 2);
    __hip_bfloat16* db    = (__hip_bfloat16*)alloc((size_t)Bb * 512 * 2);
    float*          logits= (float*)alloc((size_t)Bb * V * 4);
    int*            order = (int*)alloc(Bb * 4);
    int*            lens_s= (int*)alloc(Bb * 4);
    (void)ws_size; (void)in_sizes; (void)n_in; (void)out_size;

    // setup
    k_sort<<<1, Bb, 0, stream>>>(mr_lengths, order, lens_s);
    k_cat_w<<<2048, 256, 0, stream>>>(W_ih_enc, W_hh_enc, Wenc);
    k_cat_w<<<2048, 256, 0, stream>>>(W_ih_dec, W_hh_dec, Wdec);
    k_cvt_bf16<<<512, 256, 0, stream>>>(W_s, Wsb, 512 * 1024);
    k_cvt_bf16<<<4096, 256, 0, stream>>>(W_y, Wyb, V * 1024);
    k_gather<<<4096, 256, 0, stream>>>(emb, mr, order, Ss, Xb);
    k_gather<<<2048, 256, 0, stream>>>(emb, ref, order, Tt, Yb);
    k_zero<<<64, 256, 0, stream>>>(hbuf, 2 * Bb * 512);
    k_zero<<<64, 256, 0, stream>>>(cbuf, 2 * Bb * 512);
    k_zero<<<32, 256, 0, stream>>>((float*)hbufb, Bb * 512);  // 2 bufs * bf16 = Bb*512 floats

    const size_t ST = (size_t)Bb * 512;  // per-buffer state stride

    // encoder: 128 sequential fused GEMM+LSTM steps
    for (int s = 0; s < Ss; ++s) {
        int p = s & 1;
        k_lstm_step<<<16, 256, 0, stream>>>(
            Xb + (size_t)s * ST, hbufb + p * ST, Wenc,
            hbuf + p * ST, cbuf + p * ST,
            hbuf + (1 - p) * ST, cbuf + (1 - p) * ST, hbufb + (1 - p) * ST,
            hs, lens_s, s);
    }
    // att1 = tanh(hs @ W_a1^T), constant across decode steps
    k_att1<<<(Bb * Ss + 255) / 256, 256, 0, stream>>>(hs, W_a1, att1);

    // decoder: 48 sequential steps; enc final state sits at parity 0
    for (int t = 0; t < Tt; ++t) {
        int p = t & 1;
        k_attn<<<Bb, 256, 0, stream>>>(hbuf + p * ST, W_a2, att1, hs, mr_lengths, ctx, ctxb);
        // d = [y_t, ctx] @ W_s^T -> bf16 : 4m x 8 ngroups = 32 waves = 4 blocks
        k_gemm_to_bf16<<<4, 256, 0, stream>>>(Yb + (size_t)t * ST, ctxb, Wsb, db, 512);
        // decoder LSTM: gates = [d, s_prev] @ Wdec^T
        k_lstm_step<<<16, 256, 0, stream>>>(
            db, hbufb + p * ST, Wdec,
            hbuf + p * ST, cbuf + p * ST,
            hbuf + (1 - p) * ST, cbuf + (1 - p) * ST, hbufb + (1 - p) * ST,
            nullptr, nullptr, 0);
        // logits = [s_new, ctx] @ W_y^T : 4m x 500 ngroups = 2000 waves = 250 WGs
        k_gemm_to_f32<<<250, 256, 0, stream>>>(hbufb + (1 - p) * ST, ctxb, Wyb, logits, V);
        // softmax over V + desorted write to d_out
        k_softmax_out<<<Bb, 256, 0, stream>>>(logits, order, out, t);
    }
}
